// StandardAttention_39676907880450
// MI455X (gfx1250) — matrix operand order repaired and checked by reading, not compiled, not run on hardware
//
#include <hip/hip_runtime.h>

typedef unsigned short u16;
typedef __attribute__((ext_vector_type(16))) __bf16 v16bf;
typedef __attribute__((ext_vector_type(8)))  float  v8f;
typedef __attribute__((ext_vector_type(4)))  float  v4f;
typedef __attribute__((ext_vector_type(8)))  unsigned short v8u;

union FragAB { v16bf bf; v8u h[2]; };

__device__ __forceinline__ u16 f2bf(float f) {
  unsigned u = __float_as_uint(f);
  u += 0x7FFFu + ((u >> 16) & 1u);            // round-to-nearest-even
  return (u16)(u >> 16);
}
__device__ __forceinline__ float bf2f(u16 h) {
  return __uint_as_float(((unsigned)h) << 16);
}

__device__ __forceinline__ v8f wmma_bf16(const FragAB& a, const FragAB& b, v8f c) {
  return __builtin_amdgcn_wmma_f32_16x16x32_bf16(false, a.bf, false, b.bf,
                                                 (short)0, c, false, false);
}

// ---------------------------------------------------------------------------
// f32 -> bf16 conversion
// ---------------------------------------------------------------------------
__global__ void k_cvt(const float* __restrict__ src, u16* __restrict__ dst, int n) {
  int i = blockIdx.x * 256 + threadIdx.x;
  if (i < n) dst[i] = f2bf(src[i]);
}

// ---------------------------------------------------------------------------
// C[M,N] = A[M,K] @ W[N,K]^T   (bf16 in, f32 accumulate, bf16 or f32 out)
// Block tile 128x128, BK=64, 256 threads = 8 waves (4x2), wave tile 32x64.
// ---------------------------------------------------------------------------
template <bool OUT_F32>
__global__ __launch_bounds__(256) void k_gemm(const u16* __restrict__ A,
                                              const u16* __restrict__ W,
                                              void* __restrict__ Out,
                                              int M, int N, int K) {
  __shared__ u16 As[128][72];   // 64 + 8 pad
  __shared__ u16 Bs[128][72];

  const int lane = threadIdx.x & 31, wv = threadIdx.x >> 5;
  const int wm = wv >> 1, wn = wv & 1;
  const int m0 = blockIdx.y * 128, n0 = blockIdx.x * 128;

  v8f acc[2][4];
#pragma unroll
  for (int i = 0; i < 2; ++i)
#pragma unroll
    for (int j = 0; j < 4; ++j)
#pragma unroll
      for (int r = 0; r < 8; ++r) acc[i][j][r] = 0.0f;

  const int row_c = threadIdx.x >> 1;        // 0..127
  const int col_c = (threadIdx.x & 1) * 32;  // 0 or 32
  const int rlo = lane & 15, kh = lane >> 4;

  for (int k0 = 0; k0 < K; k0 += 64) {
    const u16* asrc = A + (size_t)(m0 + row_c) * K + k0 + col_c;
    const u16* bsrc = W + (size_t)(n0 + row_c) * K + k0 + col_c;
#pragma unroll
    for (int i = 0; i < 4; ++i) {
      *(v8u*)(&As[row_c][col_c + 8 * i]) = *(const v8u*)(asrc + 8 * i);
      *(v8u*)(&Bs[row_c][col_c + 8 * i]) = *(const v8u*)(bsrc + 8 * i);
    }
    if (k0 + 64 < K) {                       // prefetch next K tile (global_prefetch_b8)
      __builtin_prefetch(asrc + 64, 0, 1);
      __builtin_prefetch(bsrc + 64, 0, 1);
    }
    __syncthreads();

#pragma unroll
    for (int ks = 0; ks < 2; ++ks) {
      FragAB af[2], bf_[4];
#pragma unroll
      for (int mi = 0; mi < 2; ++mi) {
        const u16* p = &As[wm * 32 + mi * 16 + rlo][ks * 32 + 8 * kh];
        af[mi].h[0] = *(const v8u*)(p);
        af[mi].h[1] = *(const v8u*)(p + 16);
      }
#pragma unroll
      for (int nj = 0; nj < 4; ++nj) {
        const u16* p = &Bs[wn * 64 + nj * 16 + rlo][ks * 32 + 8 * kh];
        bf_[nj].h[0] = *(const v8u*)(p);
        bf_[nj].h[1] = *(const v8u*)(p + 16);
      }
#pragma unroll
      for (int mi = 0; mi < 2; ++mi)
#pragma unroll
        for (int nj = 0; nj < 4; ++nj)
          acc[mi][nj] = wmma_bf16(af[mi], bf_[nj], acc[mi][nj]);
    }
    __syncthreads();
  }

#pragma unroll
  for (int mi = 0; mi < 2; ++mi)
#pragma unroll
    for (int nj = 0; nj < 4; ++nj)
#pragma unroll
      for (int r = 0; r < 8; ++r) {
        size_t row = (size_t)(m0 + wm * 32 + mi * 16 + r + 8 * kh);
        size_t col = (size_t)(n0 + wn * 64 + nj * 16 + rlo);
        if (OUT_F32) ((float*)Out)[row * N + col] = acc[mi][nj][r];
        else         ((u16*)Out)[row * N + col]   = f2bf(acc[mi][nj][r]);
      }
}

// ---------------------------------------------------------------------------
// RoPE on Q,K + reshape to [B,nh,S,hd]; V transposed to [B,nh,hd,S].
// One thread per (b,s,h,d) with d<64 handles the (d, d+64) rotation pair.
// ---------------------------------------------------------------------------
__global__ void k_rope(const u16* __restrict__ Qb, const u16* __restrict__ Kb,
                       const u16* __restrict__ Vb, const float* __restrict__ cosT,
                       const float* __restrict__ sinT, u16* __restrict__ Qh,
                       u16* __restrict__ Kh, u16* __restrict__ Vt) {
  int t = blockIdx.x * 256 + threadIdx.x;      // [0, 2*2048*16*64)
  int d = t & 63;  t >>= 6;
  int h = t & 15;  t >>= 4;
  int s = t & 2047;
  int b = t >> 11;

  size_t inBase = ((size_t)(b * 2048 + s)) * 2048 + h * 128;
  float c  = cosT[s * 128 + d];
  float sn = sinT[s * 128 + d];

  float q0 = bf2f(Qb[inBase + d]),      q1 = bf2f(Qb[inBase + d + 64]);
  float k0 = bf2f(Kb[inBase + d]),      k1 = bf2f(Kb[inBase + d + 64]);
  float v0 = bf2f(Vb[inBase + d]),      v1 = bf2f(Vb[inBase + d + 64]);

  float q0r = q0 * c - q1 * sn, q1r = q1 * c + q0 * sn;
  float k0r = k0 * c - k1 * sn, k1r = k1 * c + k0 * sn;

  size_t hBase = ((size_t)(b * 16 + h) * 2048 + s) * 128;
  Qh[hBase + d]      = f2bf(q0r);
  Qh[hBase + d + 64] = f2bf(q1r);
  Kh[hBase + d]      = f2bf(k0r);
  Kh[hBase + d + 64] = f2bf(k1r);

  size_t vtBase = ((size_t)(b * 16 + h)) * 128 * 2048 + s;
  Vt[vtBase + (size_t)d * 2048]        = f2bf(v0);
  Vt[vtBase + (size_t)(d + 64) * 2048] = f2bf(v1);
}

// ---------------------------------------------------------------------------
// Fused flash attention: one workgroup = (b,h, 64 query rows); 4 waves x 16 rows.
// Streams 64-key blocks through LDS, online softmax, PV accumulated in f32.
// Mask tile is staged through LDS so the WMMA stream never waits on VMEM.
// ---------------------------------------------------------------------------
__global__ __launch_bounds__(128) void k_attn(const u16* __restrict__ Qh,
                                              const u16* __restrict__ Kh,
                                              const u16* __restrict__ Vt,
                                              const float* __restrict__ mask,
                                              u16* __restrict__ AO) {
  __shared__ u16  Ks[64][136];      // keys x hd
  __shared__ u16  Vs[128][72];      // hd x keys
  __shared__ u16  Ps[4][16][72];    // per-wave P tile (D-layout -> A-layout bounce)
  __shared__ float Ms[64][68];      // mask tile: 64 q-rows x 64 k-cols (+pad)

  const int lane = threadIdx.x & 31, wave = threadIdx.x >> 5;
  const int bh = blockIdx.y, b = bh >> 4, h = bh & 15;
  const int qblk = blockIdx.x * 64;
  const int q0 = qblk + wave * 16;
  const int nlo = lane & 15, kh = lane >> 4;
  const float scale = 0.08838834764831845f;    // 1/sqrt(128)

  // Q fragments for this wave's 16 rows, kept in registers for the whole pass.
  FragAB qf[4];
  {
    const u16* qp = Qh + ((size_t)bh * 2048 + (q0 + nlo)) * 128;
#pragma unroll
    for (int ks = 0; ks < 4; ++ks) {
      qf[ks].h[0] = *(const v8u*)(qp + ks * 32 + 8 * kh);
      qf[ks].h[1] = *(const v8u*)(qp + ks * 32 + 8 * kh + 16);
    }
  }

  v8f of[8];
  float m_run[8], l_run[8];
#pragma unroll
  for (int j = 0; j < 8; ++j)
#pragma unroll
    for (int r = 0; r < 8; ++r) of[j][r] = 0.0f;
#pragma unroll
  for (int r = 0; r < 8; ++r) { m_run[r] = -1e30f; l_run[r] = 0.0f; }

  const u16* Kbase = Kh + (size_t)bh * 2048 * 128;
  const u16* Vbase = Vt + (size_t)bh * 128 * 2048;

  for (int kb = 0; kb < 2048; kb += 64) {
    // Cooperative stage of K (64x128), V^T (128x64), mask (64x64 f32) into LDS.
    {
      int t = threadIdx.x;
      int krow = t >> 1, c0 = (t & 1) * 64;
      const u16* ksrc = Kbase + (size_t)(kb + krow) * 128 + c0;
      u16* kdst = &Ks[krow][c0];
#pragma unroll
      for (int i = 0; i < 8; ++i) *(v8u*)(kdst + 8 * i) = *(const v8u*)(ksrc + 8 * i);

      const u16* vsrc = Vbase + (size_t)t * 2048 + kb;
      u16* vdst = &Vs[t][0];
#pragma unroll
      for (int i = 0; i < 8; ++i) *(v8u*)(vdst + 8 * i) = *(const v8u*)(vsrc + 8 * i);

      // mask row (qblk+krow), 32 floats per thread
      int mc0 = (t & 1) * 32;
      const float* msrc = mask + (size_t)(qblk + krow) * 2048 + kb + mc0;
      float* mdst = &Ms[krow][mc0];
#pragma unroll
      for (int i = 0; i < 8; ++i) *(v4f*)(mdst + 4 * i) = *(const v4f*)(msrc + 4 * i);

      if (kb + 64 < 2048) {                  // prefetch next key block
        __builtin_prefetch(ksrc + 64 * 128, 0, 1);
        __builtin_prefetch(vsrc + 64, 0, 1);
      }
    }
    __syncthreads();

    // GEMM1: S = Q @ K^T for 64 keys -> 4 score fragments (16x16 each).
    v8f sc[4];
#pragma unroll
    for (int nt = 0; nt < 4; ++nt) {
      v8f a;
#pragma unroll
      for (int r = 0; r < 8; ++r) a[r] = 0.0f;
      const u16* krow = &Ks[nt * 16 + nlo][0];
#pragma unroll
      for (int ks = 0; ks < 4; ++ks) {
        FragAB bf_;
        bf_.h[0] = *(const v8u*)(krow + ks * 32 + 8 * kh);
        bf_.h[1] = *(const v8u*)(krow + ks * 32 + 8 * kh + 16);
        a = wmma_bf16(qf[ks], bf_, a);
      }
      sc[nt] = a;
    }

    // Add attention mask from LDS (broadcast over b,h).
#pragma unroll
    for (int nt = 0; nt < 4; ++nt)
#pragma unroll
      for (int r = 0; r < 8; ++r)
        sc[nt][r] += Ms[wave * 16 + r + 8 * kh][nt * 16 + nlo];

    // Online softmax update (row reductions across the 16-lane N-groups).
    float alpha[8];
#pragma unroll
    for (int r = 0; r < 8; ++r) {
      float mx = fmaxf(fmaxf(sc[0][r], sc[1][r]), fmaxf(sc[2][r], sc[3][r]));
      mx = fmaxf(mx, __shfl_xor(mx, 1, 32));
      mx = fmaxf(mx, __shfl_xor(mx, 2, 32));
      mx = fmaxf(mx, __shfl_xor(mx, 4, 32));
      mx = fmaxf(mx, __shfl_xor(mx, 8, 32));
      float mnew = fmaxf(m_run[r], mx);
      alpha[r] = __expf((m_run[r] - mnew) * scale);
      float lsum = 0.0f;
#pragma unroll
      for (int nt = 0; nt < 4; ++nt) {
        float p = __expf((sc[nt][r] - mnew) * scale);
        sc[nt][r] = p;
        lsum += p;
      }
      lsum += __shfl_xor(lsum, 1, 32);
      lsum += __shfl_xor(lsum, 2, 32);
      lsum += __shfl_xor(lsum, 4, 32);
      lsum += __shfl_xor(lsum, 8, 32);
      l_run[r] = l_run[r] * alpha[r] + lsum;
      m_run[r] = mnew;
    }
#pragma unroll
    for (int j = 0; j < 8; ++j)
#pragma unroll
      for (int r = 0; r < 8; ++r) of[j][r] *= alpha[r];

    // P: D-fragment layout -> A-fragment layout via per-wave LDS bounce.
#pragma unroll
    for (int nt = 0; nt < 4; ++nt)
#pragma unroll
      for (int r = 0; r < 8; ++r)
        Ps[wave][r + 8 * kh][nt * 16 + nlo] = f2bf(sc[nt][r]);
    asm volatile("s_wait_dscnt 0" ::: "memory");   // wave-internal store->load ordering

    // GEMM2: O += P(16x64) @ V(64x128).
#pragma unroll
    for (int ks2 = 0; ks2 < 2; ++ks2) {
      FragAB pa;
      const u16* prow = &Ps[wave][nlo][0];
      pa.h[0] = *(const v8u*)(prow + ks2 * 32 + 8 * kh);
      pa.h[1] = *(const v8u*)(prow + ks2 * 32 + 8 * kh + 16);
#pragma unroll
      for (int j = 0; j < 8; ++j) {
        FragAB vb_;
        const u16* vrow = &Vs[j * 16 + nlo][0];
        vb_.h[0] = *(const v8u*)(vrow + ks2 * 32 + 8 * kh);
        vb_.h[1] = *(const v8u*)(vrow + ks2 * 32 + 8 * kh + 16);
        of[j] = wmma_bf16(pa, vb_, of[j]);
      }
    }
    __syncthreads();
  }

  // Normalize and store in [B*S, H] layout for the output projection GEMM.
  float inv[8];
#pragma unroll
  for (int r = 0; r < 8; ++r) inv[r] = 1.0f / l_run[r];
#pragma unroll
  for (int j = 0; j < 8; ++j)
#pragma unroll
    for (int r = 0; r < 8; ++r) {
      size_t row = (size_t)(b * 2048 + q0 + r + 8 * kh);
      size_t col = (size_t)(h * 128 + j * 16 + nlo);
      AO[row * 2048 + col] = f2bf(of[j][r] * inv[r]);
    }
}

// ---------------------------------------------------------------------------
extern "C" void kernel_launch(void* const* d_in, const int* in_sizes, int n_in,
                              void* d_out, int out_size, void* d_ws, size_t ws_size,
                              hipStream_t stream) {
  const float* hs   = (const float*)d_in[0];
  const float* cosT = (const float*)d_in[1];
  const float* sinT = (const float*)d_in[2];
  const float* mask = (const float*)d_in[3];
  const float* Wq   = (const float*)d_in[4];
  const float* Wk   = (const float*)d_in[5];
  const float* Wv   = (const float*)d_in[6];
  const float* Wo   = (const float*)d_in[7];

  const size_t MB = 1024 * 1024;
  char* ws = (char*)d_ws;
  u16* XB  = (u16*)(ws + 0 * MB);     // X bf16      [4096,2048]  16MB
  u16* WQb = (u16*)(ws + 16 * MB);    // Wq bf16     [2048,2048]   8MB
  u16* WKb = (u16*)(ws + 24 * MB);
  u16* WVb = (u16*)(ws + 32 * MB);
  u16* WOb = (u16*)(ws + 40 * MB);
  u16* QB  = (u16*)(ws + 48 * MB);    // Q bf16 [B,S,H]           16MB
  u16* KB  = (u16*)(ws + 64 * MB);
  u16* VB  = (u16*)(ws + 80 * MB);
  u16* QH  = (u16*)(ws + 96 * MB);    // Q head-major [B,nh,S,hd] 16MB
  u16* KH  = (u16*)(ws + 112 * MB);
  u16* VT  = (u16*)(ws + 128 * MB);   // V transposed [B,nh,hd,S] 16MB
  u16* AO  = (u16*)(ws + 144 * MB);   // attn out [B*S,H] bf16    16MB

  const int NX = 2 * 2048 * 2048;     // 8388608
  const int NW = 2048 * 2048;         // 4194304

  k_cvt<<<(NX + 255) / 256, 256, 0, stream>>>(hs, XB, NX);
  k_cvt<<<(NW + 255) / 256, 256, 0, stream>>>(Wq, WQb, NW);
  k_cvt<<<(NW + 255) / 256, 256, 0, stream>>>(Wk, WKb, NW);
  k_cvt<<<(NW + 255) / 256, 256, 0, stream>>>(Wv, WVb, NW);
  k_cvt<<<(NW + 255) / 256, 256, 0, stream>>>(Wo, WOb, NW);

  dim3 ggrid(2048 / 128, 4096 / 128);  // (N/128, M/128)
  k_gemm<false><<<ggrid, 256, 0, stream>>>(XB, WQb, QB, 4096, 2048, 2048);
  k_gemm<false><<<ggrid, 256, 0, stream>>>(XB, WKb, KB, 4096, 2048, 2048);
  k_gemm<false><<<ggrid, 256, 0, stream>>>(XB, WVb, VB, 4096, 2048, 2048);

  k_rope<<<(2 * 2048 * 16 * 64) / 256, 256, 0, stream>>>(QB, KB, VB, cosT, sinT,
                                                         QH, KH, VT);

  dim3 agrid(2048 / 64, 2 * 16);       // (q-tiles, B*nh)
  k_attn<<<agrid, 128, 0, stream>>>(QH, KH, VT, mask, AO);

  k_gemm<true><<<ggrid, 256, 0, stream>>>(AO, WOb, d_out, 4096, 2048, 2048);
}
